// NIC_89120571392584
// MI455X (gfx1250) — compile-verified, matrix operand order checked by reading, not run on hardware
//
#include <hip/hip_runtime.h>
#include <hip/hip_bf16.h>
#include <math.h>

// ---------------------------------------------------------------------------
// Problem constants (from reference)
// ---------------------------------------------------------------------------
#define BSZ 64
#define GG  360
#define VV  100
#define DD  32
#define UU  512
#define TT  15
#define VOC 5000
#define TDD 512
#define HH  8
#define EE  544           // D + U
#define HD  68            // E / H
#define QKN 1088          // q+k output columns
#define BG  (BSZ*GG)      // 23040
#define NEGS 0.2f

typedef __bf16 bf16;
typedef float  v8f   __attribute__((ext_vector_type(8)));
typedef bf16   v16bf __attribute__((ext_vector_type(16)));
typedef bf16   v4bf  __attribute__((ext_vector_type(4)));

// ---------------------------------------------------------------------------
// bf16 <-> f32 via bit punning (round-to-nearest-even on pack)
// ---------------------------------------------------------------------------
__device__ __forceinline__ bf16 f2bf(float f) {
    union { float f; unsigned u; } x; x.f = f;
    unsigned r = (x.u + 0x7FFFu + ((x.u >> 16) & 1u)) >> 16;
    union { unsigned short s; bf16 b; } y; y.s = (unsigned short)r;
    return y.b;
}
__device__ __forceinline__ float bf2f(bf16 b) {
    union { unsigned short s; bf16 b; } y; y.b = b;
    union { unsigned u; float f; } x; x.u = ((unsigned)y.s) << 16;
    return x.f;
}
__device__ __forceinline__ float leaky(float v) { return v > 0.f ? v : NEGS * v; }

// ---------------------------------------------------------------------------
// Wave-level 16x16 bf16 WMMA tile:  C(16x16,f32) += A(16xK) * Bt(16xK)^T
// A: row-major M x K (lda), Bt: row-major N x K (ldb)  (B stored as W[N][K]).
// Per ISA 7.12.2 16-bit A layout: lanes 0-15 hold M=lane, K in {0..7,16..23};
// lanes 16-31 hold M=lane-16, K in {8..15,24..31}. B mirrors A with M<->N.
//
// Branch-free: out-of-range tile rows are CLAMPED to a valid row (their
// garbage only lands in output rows/cols that store_tile masks). Main K loop
// is guard-free with 8-byte vector loads (all operands are 8B-aligned by
// construction); only the final partial K-step (K%32 != 0) uses clamped-index
// + value-select (cndmask, still no exec divergence).
// ---------------------------------------------------------------------------
__device__ __forceinline__ v8f wmma_tile(const bf16* __restrict__ A, int lda,
                                         const bf16* __restrict__ Bt, int ldb,
                                         int M0, int N0, int K, int Mlim, int Nlim) {
    const int lane = threadIdx.x & 31;
    const int r    = lane & 15;
    const int kg   = (lane >> 4) * 8;
    int ar = M0 + r; ar = (ar < Mlim) ? ar : (Mlim - 1);
    int br = N0 + r; br = (br < Nlim) ? br : (Nlim - 1);
    const bf16* Ap = A  + (size_t)ar * lda + kg;
    const bf16* Bp = Bt + (size_t)br * ldb + kg;
    v8f acc = {0.f,0.f,0.f,0.f,0.f,0.f,0.f,0.f};
    const int Kfull = K & ~31;
    int k0 = 0;
    for (; k0 < Kfull; k0 += 32) {
        __builtin_prefetch(Ap + k0 + 64, 0, 1);   // global_prefetch_b8
        __builtin_prefetch(Bp + k0 + 64, 0, 1);
        union { v16bf v; v4bf q[4]; } ua, ub;
        ua.q[0] = *(const v4bf*)(Ap + k0);
        ua.q[1] = *(const v4bf*)(Ap + k0 + 4);
        ua.q[2] = *(const v4bf*)(Ap + k0 + 16);
        ua.q[3] = *(const v4bf*)(Ap + k0 + 20);
        ub.q[0] = *(const v4bf*)(Bp + k0);
        ub.q[1] = *(const v4bf*)(Bp + k0 + 4);
        ub.q[2] = *(const v4bf*)(Bp + k0 + 16);
        ub.q[3] = *(const v4bf*)(Bp + k0 + 20);
        acc = __builtin_amdgcn_wmma_f32_16x16x32_bf16(
                  false, ua.v, false, ub.v, (short)0, acc, false, false);
    }
    if (k0 < K) {                                  // branchless K tail
        const bf16 bz = f2bf(0.f);
        const bf16* Ar = A  + (size_t)ar * lda;
        const bf16* Br = Bt + (size_t)br * ldb;
        v16bf av, bv;
#pragma unroll
        for (int i = 0; i < 8; ++i) {
            int ka = k0 + kg + i;
            int kb = ka + 16;
            int kac = (ka < K) ? ka : 0;
            int kbc = (kb < K) ? kb : 0;
            bf16 a0 = Ar[kac], a1 = Ar[kbc];
            bf16 b0 = Br[kac], b1 = Br[kbc];
            av[i]     = (ka < K) ? a0 : bz;
            av[8 + i] = (kb < K) ? a1 : bz;
            bv[i]     = (ka < K) ? b0 : bz;
            bv[8 + i] = (kb < K) ? b1 : bz;
        }
        acc = __builtin_amdgcn_wmma_f32_16x16x32_bf16(
                  false, av, false, bv, (short)0, acc, false, false);
    }
    return acc;
}

// C/D layout: VGPR j, lanes 0-15 -> (M=j, N=lane); lanes 16-31 -> (M=8+j, N=lane-16)
__device__ __forceinline__ void store_tile(float* C, int ldc, bf16* Cbf, int ldcb,
                                           int M0, int N0, int Mlim, int Nlim,
                                           v8f acc, const float* bias, int act) {
    const int lane = threadIdx.x & 31;
    const int n = N0 + (lane & 15);
    const int mb = M0 + (lane >> 4) * 8;
    if (n >= Nlim) return;
    const float bv = bias ? bias[n] : 0.f;
#pragma unroll
    for (int j = 0; j < 8; ++j) {
        int m = mb + j;
        if (m < Mlim) {
            float v = acc[j] + bv;
            if (act == 1) v = leaky(v);
            if (C)   C[(size_t)m * ldc + n] = v;
            if (Cbf) Cbf[(size_t)m * ldcb + n] = f2bf(v);
        }
    }
}

// ---------------------------------------------------------------------------
// Generic batched bf16 GEMM:  C[z] = act(A[z] * Bt[z]^T + bias)
// one 16x16 output tile per wave; 8 waves per block
// ---------------------------------------------------------------------------
__global__ __launch_bounds__(256) void k_gemm(
        const bf16* __restrict__ A, int lda, long long sA,
        const bf16* __restrict__ Bt, int ldb, long long sB,
        const float* __restrict__ bias,
        float* C, int ldc, long long sC,
        bf16* Cbf, int ldcb, long long sCb,
        int M, int N, int K, int act) {
    const int z = blockIdx.z;
    A  += (size_t)z * sA;
    Bt += (size_t)z * sB;
    if (C)   C   += (size_t)z * sC;
    if (Cbf) Cbf += (size_t)z * sCb;
    const int mt = (M + 15) >> 4, nt = (N + 15) >> 4;
    const int wave = blockIdx.x * (blockDim.x >> 5) + (threadIdx.x >> 5);
    if (wave >= mt * nt) return;           // whole-wave exit: EXEC stays full
    const int mi = wave / nt, ni = wave % nt;
    v8f acc = wmma_tile(A, lda, Bt, ldb, mi * 16, ni * 16, K, M, N);
    store_tile(C, ldc, Cbf, ldcb, mi * 16, ni * 16, M, N, acc, bias, act);
}

// ---------------------------------------------------------------------------
// S0/E0:  per (b,h):  E0[q,k] = exp( scale * qf[b,q,h,:] . kf[b,k,h,:] ), bf16
// ---------------------------------------------------------------------------
__global__ __launch_bounds__(256) void k_s0e0(const bf16* __restrict__ qk,
                                              bf16* __restrict__ E0, float scale) {
    const int z = blockIdx.z;                 // z = b*8 + h
    const int b = z >> 3, h = z & 7;
    const bf16* A  = qk + (size_t)b * GG * QKN + h * HD;          // qf rows (lda=QKN)
    const bf16* Bt = A + EE;                                      // kf rows
    const int nt = (GG + 15) >> 4;                                // 23
    const int wave = blockIdx.x * (blockDim.x >> 5) + (threadIdx.x >> 5);
    if (wave >= nt * nt) return;
    const int mi = wave / nt, ni = wave % nt;
    v8f acc = wmma_tile(A, QKN, Bt, QKN, mi * 16, ni * 16, HD, GG, GG);
    const int lane = threadIdx.x & 31;
    const int n = ni * 16 + (lane & 15);
    const int mb = mi * 16 + (lane >> 4) * 8;
    if (n >= GG) return;
    bf16* out = E0 + (size_t)z * GG * GG;
#pragma unroll
    for (int j = 0; j < 8; ++j) {
        int m = mb + j;
        if (m < GG) out[(size_t)m * GG + n] = f2bf(__expf(acc[j] * scale));
    }
}

// ---------------------------------------------------------------------------
// small prep kernels
// ---------------------------------------------------------------------------
__global__ void k_cvt(const float* __restrict__ src, bf16* __restrict__ dst, long long n) {
    long long i = (long long)blockIdx.x * blockDim.x + threadIdx.x;
    const long long st = (long long)gridDim.x * blockDim.x;
    for (; i < n; i += st) dst[i] = f2bf(src[i]);
}

__global__ void k_packwg(const float* __restrict__ Wih, const float* __restrict__ Whh,
                         const float* __restrict__ bih, const float* __restrict__ bhh,
                         bf16* __restrict__ Wg, float* __restrict__ biasg) {
    const long long tot = (long long)(4 * UU) * (DD + TDD + UU);  // 2048 x 1056
    long long i = (long long)blockIdx.x * blockDim.x + threadIdx.x;
    const long long st = (long long)gridDim.x * blockDim.x;
    for (; i < tot; i += st) {
        int n = (int)(i / 1056), j = (int)(i % 1056);
        float v = (j < EE) ? Wih[(size_t)n * EE + j] : Whh[(size_t)n * UU + (j - EE)];
        Wg[i] = f2bf(v);
        if (j == 0) biasg[n] = bih[n] + bhh[n];
    }
}

__global__ void k_init(const float* __restrict__ a0, const float* __restrict__ c0,
                       float* a, float* c, bf16* a_bf) {
    int i = blockIdx.x * blockDim.x + threadIdx.x;
    if (i < BSZ * UU) { a[i] = a0[i]; c[i] = c0[i]; a_bf[i] = f2bf(a0[i]); }
}

// encoder post: per (b,g) one wave: bias + LN(D=32) + leaky; writes f32 + bf16
__global__ void k_encpost(float* __restrict__ feat, bf16* __restrict__ feat_bf,
                          const float* __restrict__ enc_b, const float* __restrict__ enc_g,
                          const float* __restrict__ enc_beta) {
    const int wid = (blockIdx.x * blockDim.x + threadIdx.x) >> 5;
    const int lane = threadIdx.x & 31;
    if (wid >= BSZ * GG) return;
    const int g = wid % GG;
    float* p = feat + (size_t)wid * DD;
    float x = p[lane] + enc_b[g * DD + lane];
    float s = x;
    for (int o = 16; o; o >>= 1) s += __shfl_xor(s, o, 32);
    const float mean = s * (1.f / 32.f);
    float d = x - mean, q = d * d;
    for (int o = 16; o; o >>= 1) q += __shfl_xor(q, o, 32);
    float y = d * rsqrtf(q * (1.f / 32.f) + 1e-5f) * enc_g[lane] + enc_beta[lane];
    y = leaky(y);
    p[lane] = y;
    feat_bf[(size_t)wid * DD + lane] = f2bf(y);
}

// ---------------------------------------------------------------------------
// per-timestep kernels
// ---------------------------------------------------------------------------
// ev[b,h,k] = exp( scale * qa[b,h,:] . kf[b,k,h,:] )
__global__ __launch_bounds__(256) void k_v_ev(const float* __restrict__ qa,
                                              const bf16* __restrict__ qk,
                                              float* __restrict__ ev, float scale) {
    const int b = blockIdx.x;
    for (int idx = threadIdx.x; idx < HH * GG; idx += blockDim.x) {
        const int h = idx / GG, k = idx - h * GG;
        const bf16* kf = qk + ((size_t)(b * GG + k)) * QKN + EE + h * HD;
        const float* q = qa + (size_t)b * EE + h * HD;
        float s = 0.f;
#pragma unroll 4
        for (int d = 0; d < HD; ++d) s += q[d] * bf2f(kf[d]);
        ev[(size_t)b * HH * GG + idx] = __expf(s * scale);
    }
}

// Per b: loop h: denom[q]=E0[q,:].ev ; colsum[k]=sum_q E0[q,k]/denom[q];
// wacc[k]+=ev[k]*colsum[k]. Then attention weight, ctx.
__global__ __launch_bounds__(512) void k_attnw(const bf16* __restrict__ E0,
                                               const float* __restrict__ ev,
                                               const float* __restrict__ feat,
                                               float* __restrict__ attn_out,
                                               float* __restrict__ ctx, int t) {
    __shared__ float invden[GG], evl[GG], wacc[GG];
    const int b = blockIdx.x, tid = threadIdx.x;
    for (int k = tid; k < GG; k += 512) wacc[k] = 0.f;
    __syncthreads();
    for (int h = 0; h < HH; ++h) {
        const bf16* E = E0 + ((size_t)(b * HH + h)) * GG * GG;
        const float* evh = ev + (size_t)(b * HH + h) * GG;
        for (int k = tid; k < GG; k += 512) evl[k] = evh[k];
        __syncthreads();
        const int wv = tid >> 5, lane = tid & 31;       // 16 waves
        for (int q = wv; q < GG; q += 16) {
            const bf16* row = E + (size_t)q * GG;
            __builtin_prefetch(row + 16 * GG, 0, 0);
            float s = 0.f;
            for (int k = lane; k < GG; k += 32) s += bf2f(row[k]) * evl[k];
            for (int o = 16; o; o >>= 1) s += __shfl_xor(s, o, 32);
            if (lane == 0) invden[q] = 1.f / s;
        }
        __syncthreads();
        for (int k = tid; k < GG; k += 512) {
            float s = 0.f;
            for (int q = 0; q < GG; ++q) s += bf2f(E[(size_t)q * GG + k]) * invden[q];
            wacc[k] += evl[k] * s;
        }
        __syncthreads();
    }
    const float inv = 1.f / (float)(HH * GG);
    for (int k = tid; k < GG; k += 512) {
        float w = wacc[k] * inv;
        wacc[k] = w;
        attn_out[((size_t)b * TT + t) * GG + k] = w;
    }
    __syncthreads();
    if (tid < DD) {
        float s = 0.f;
        for (int k = 0; k < GG; ++k) s += wacc[k] * feat[((size_t)(b * GG + k)) * DD + tid];
        ctx[b * DD + tid] = s;
    }
}

// Abuf[b, 0:32]=ctx | [32:544]=emb[text[b,t]] | [544:1056]=a   (bf16)
__global__ void k_abuf(const float* __restrict__ ctx, const float* __restrict__ emb,
                       const int* __restrict__ text, const float* __restrict__ a,
                       bf16* __restrict__ Abuf, int t) {
    int i = blockIdx.x * blockDim.x + threadIdx.x;
    if (i >= BSZ * (DD + TDD + UU)) return;
    int b = i / 1056, j = i % 1056;
    float v;
    if (j < DD)          v = ctx[b * DD + j];
    else if (j < DD+TDD) v = emb[(size_t)text[b * TT + t] * TDD + (j - DD)];
    else                 v = a[b * UU + (j - DD - TDD)];
    Abuf[i] = f2bf(v);
}

// LSTM cell + LayerNorm(U=512) + leaky; writes c, a(f32+bf16), s(bf16)
__global__ __launch_bounds__(512) void k_lstm(const float* __restrict__ gates,
                                              float* __restrict__ c, float* __restrict__ a,
                                              bf16* __restrict__ a_bf, bf16* __restrict__ s_bf,
                                              const float* __restrict__ ln_g,
                                              const float* __restrict__ ln_b) {
    __shared__ float red[16];
    const int b = blockIdx.x, u = threadIdx.x;
    const float* g4 = gates + (size_t)b * 4 * UU;
    const float iv = g4[u], fv = g4[UU + u], gv = g4[2 * UU + u], ov = g4[3 * UU + u];
    const float si = 1.f / (1.f + __expf(-iv));
    const float sf = 1.f / (1.f + __expf(-fv));
    const float so = 1.f / (1.f + __expf(-ov));
    const float cn = sf * c[b * UU + u] + si * tanhf(gv);
    c[b * UU + u] = cn;
    const float h = so * tanhf(cn);
    float s = h;
    for (int o = 16; o; o >>= 1) s += __shfl_xor(s, o, 32);
    if ((u & 31) == 0) red[u >> 5] = s;
    __syncthreads();
    float tot = 0.f;
    for (int i = 0; i < 16; ++i) tot += red[i];
    const float mean = tot * (1.f / (float)UU);
    __syncthreads();
    const float d = h - mean;
    float q = d * d;
    for (int o = 16; o; o >>= 1) q += __shfl_xor(q, o, 32);
    if ((u & 31) == 0) red[u >> 5] = q;
    __syncthreads();
    float v2 = 0.f;
    for (int i = 0; i < 16; ++i) v2 += red[i];
    const float an = d * rsqrtf(v2 * (1.f / (float)UU) + 1e-5f) * ln_g[u] + ln_b[u];
    a[b * UU + u] = an;
    a_bf[b * UU + u] = f2bf(an);
    s_bf[b * UU + u] = f2bf(leaky(an));
}

__global__ __launch_bounds__(256) void k_softmax(const float* __restrict__ logits,
                                                 float* __restrict__ out, int t) {
    __shared__ float red[8];
    const int b = blockIdx.x, tid = threadIdx.x;
    const float* L = logits + (size_t)b * VOC;
    float m = -1e30f;
    for (int i = tid; i < VOC; i += 256) m = fmaxf(m, L[i]);
    for (int o = 16; o; o >>= 1) m = fmaxf(m, __shfl_xor(m, o, 32));
    if ((tid & 31) == 0) red[tid >> 5] = m;
    __syncthreads();
    float mm = -1e30f;
    for (int i = 0; i < 8; ++i) mm = fmaxf(mm, red[i]);
    __syncthreads();
    float s = 0.f;
    for (int i = tid; i < VOC; i += 256) s += __expf(L[i] - mm);
    for (int o = 16; o; o >>= 1) s += __shfl_xor(s, o, 32);
    if ((tid & 31) == 0) red[tid >> 5] = s;
    __syncthreads();
    float ss = 0.f;
    for (int i = 0; i < 8; ++i) ss += red[i];
    const float inv = 1.f / ss;
    float* O = out + ((size_t)b * TT + t) * VOC;
    for (int i = tid; i < VOC; i += 256) O[i] = __expf(L[i] - mm) * inv;
}

// ---------------------------------------------------------------------------
// host side
// ---------------------------------------------------------------------------
static inline int cdiv(long long a, long long b) { return (int)((a + b - 1) / b); }

extern "C" void kernel_launch(void* const* d_in, const int* in_sizes, int n_in,
                              void* d_out, int out_size, void* d_ws, size_t ws_size,
                              hipStream_t stream) {
    const float* features = (const float*)d_in[0];
    const int*   text     = (const int*)  d_in[1];
    const float* a0       = (const float*)d_in[2];
    const float* c0       = (const float*)d_in[3];
    const float* enc_W    = (const float*)d_in[4];
    const float* enc_b    = (const float*)d_in[5];
    const float* enc_g    = (const float*)d_in[6];
    const float* enc_beta = (const float*)d_in[7];
    const float* emb      = (const float*)d_in[8];
    const float* in_w     = (const float*)d_in[9];
    const float* in_b     = (const float*)d_in[10];
    const float* Wih      = (const float*)d_in[11];
    const float* Whh      = (const float*)d_in[12];
    const float* bih      = (const float*)d_in[13];
    const float* bhh      = (const float*)d_in[14];
    const float* ln_g     = (const float*)d_in[15];
    const float* ln_b     = (const float*)d_in[16];
    const float* W1       = (const float*)d_in[17];
    const float* b1       = (const float*)d_in[18];
    const float* W2       = (const float*)d_in[19];
    const float* b2       = (const float*)d_in[20];

    float* out_probs = (float*)d_out;                              // (64,15,5000)
    float* out_attn  = out_probs + (size_t)BSZ * TT * VOC;         // (64,15,360,1)

    char* ws = (char*)d_ws;
    size_t cur = 0;
    auto alloc = [&](size_t bytes) -> void* {
        void* p = ws + cur;
        cur = (cur + bytes + 255) & ~(size_t)255;
        return p;
    };
    bf16*  feats_bf = (bf16*)alloc((size_t)BSZ * GG * VV * 2);
    bf16*  encW_bf  = (bf16*)alloc((size_t)GG * DD * VV * 2);
    bf16*  inw_bf   = (bf16*)alloc((size_t)QKN * EE * 2);
    bf16*  Wg_bf    = (bf16*)alloc((size_t)4 * UU * (DD + TDD + UU) * 2);
    float* biasg    = (float*)alloc((size_t)4 * UU * 4);
    bf16*  W1_bf    = (bf16*)alloc((size_t)256 * UU * 2);
    bf16*  W2_bf    = (bf16*)alloc((size_t)VOC * 256 * 2);
    float* feat     = (float*)alloc((size_t)BG * DD * 4);
    bf16*  feat_bf  = (bf16*)alloc((size_t)BG * DD * 2);
    bf16*  qk_bf    = (bf16*)alloc((size_t)BG * QKN * 2);
    bf16*  E0       = (bf16*)alloc((size_t)BSZ * HH * GG * GG * 2);
    float* a_st     = (float*)alloc((size_t)BSZ * UU * 4);
    float* c_st     = (float*)alloc((size_t)BSZ * UU * 4);
    bf16*  a_bf     = (bf16*)alloc((size_t)BSZ * UU * 2);
    bf16*  s_bf     = (bf16*)alloc((size_t)BSZ * UU * 2);
    float* qa       = (float*)alloc((size_t)BSZ * EE * 4);
    float* ev       = (float*)alloc((size_t)BSZ * HH * GG * 4);
    float* ctx      = (float*)alloc((size_t)BSZ * DD * 4);
    bf16*  Abuf     = (bf16*)alloc((size_t)BSZ * 1056 * 2);
    float* gates    = (float*)alloc((size_t)BSZ * 4 * UU * 4);
    bf16*  h1_bf    = (bf16*)alloc((size_t)BSZ * 256 * 2);
    float* logits   = (float*)alloc((size_t)BSZ * VOC * 4);
    (void)ws_size; (void)n_in; (void)in_sizes; (void)out_size;

    const float scale = 1.0f / sqrtf((float)HD);

    auto gemm = [&](const bf16* A, int lda, long long sA,
                    const bf16* Bt, int ldb, long long sB,
                    const float* bias, float* C, int ldc, long long sC,
                    bf16* Cbf, int ldcb, long long sCb,
                    int M, int N, int K, int act, int Z) {
        int waves = ((M + 15) >> 4) * ((N + 15) >> 4);
        dim3 grid(cdiv(waves, 8), 1, Z);
        k_gemm<<<grid, 256, 0, stream>>>(A, lda, sA, Bt, ldb, sB, bias,
                                         C, ldc, sC, Cbf, ldcb, sCb, M, N, K, act);
    };

    // ---- one-time precompute --------------------------------------------
    k_cvt<<<4096, 256, 0, stream>>>(features, feats_bf, (long long)BSZ * GG * VV);
    k_cvt<<<4096, 256, 0, stream>>>(enc_W, encW_bf, (long long)GG * DD * VV);
    k_cvt<<<2048, 256, 0, stream>>>(in_w, inw_bf, (long long)QKN * EE);
    k_cvt<<<512,  256, 0, stream>>>(W1, W1_bf, (long long)256 * UU);
    k_cvt<<<4096, 256, 0, stream>>>(W2, W2_bf, (long long)VOC * 256);
    k_packwg<<<4096, 256, 0, stream>>>(Wih, Whh, bih, bhh, Wg_bf, biasg);
    k_init<<<cdiv(BSZ * UU, 256), 256, 0, stream>>>(a0, c0, a_st, c_st, a_bf);

    // encoder: per-g batched GEMM (64x32x100) -> feat (pre-LN)
    gemm(feats_bf, GG * VV, VV,           // A: features[b, g*100 + v]
         encW_bf, VV, (long long)DD * VV, // Bt: enc_W[g] (32 x 100)
         nullptr, feat, GG * DD, DD,      // C: feat[b,g,d], ldc = 360*32
         nullptr, 0, 0, BSZ, DD, VV, 0, GG);
    k_encpost<<<cdiv(BSZ * GG * 32, 256), 256, 0, stream>>>(feat, feat_bf, enc_b, enc_g, enc_beta);

    // qf|kf = feat @ [Wq_feat|Wk_feat]^T + [bq|bk]   (23040 x 1088, K=32)
    gemm(feat_bf, DD, 0, inw_bf, EE, 0, in_b,
         nullptr, 0, 0, qk_bf, QKN, 0,
         BG, QKN, DD, 0, 1);

    // E0 = exp(scale * qf . kf)  per (b,h)
    {
        int waves = 23 * 23;
        dim3 grid(cdiv(waves, 8), 1, BSZ * HH);
        k_s0e0<<<grid, 256, 0, stream>>>(qk_bf, E0, scale);
    }

    // ---- timestep loop ---------------------------------------------------
    for (int t = 0; t < TT; ++t) {
        // qa = a @ Wq_a^T   (64 x 544, K=512);  Wq_a = in_w[0:544, 32:544]
        gemm(a_bf, UU, 0, inw_bf + DD, EE, 0, nullptr,
             qa, EE, 0, nullptr, 0, 0, BSZ, EE, UU, 0, 1);

        k_v_ev<<<BSZ, 256, 0, stream>>>(qa, qk_bf, ev, scale);
        k_attnw<<<BSZ, 512, 0, stream>>>(E0, ev, feat, out_attn, ctx, t);
        k_abuf<<<cdiv(BSZ * 1056, 256), 256, 0, stream>>>(ctx, emb, text, a_st, Abuf, t);

        // gates = [ctx|x_t|a] @ [Wih|Whh]^T + (bih+bhh)   (64 x 2048, K=1056)
        gemm(Abuf, 1056, 0, Wg_bf, 1056, 0, biasg,
             gates, 4 * UU, 0, nullptr, 0, 0, BSZ, 4 * UU, 1056, 0, 1);

        k_lstm<<<BSZ, UU, 0, stream>>>(gates, c_st, a_st, a_bf, s_bf, ln_g, ln_b);

        // h1 = leaky(s @ W1^T + b1)  (64 x 256, K=512) -> bf16
        gemm(s_bf, UU, 0, W1_bf, UU, 0, b1,
             nullptr, 0, 0, h1_bf, 256, 0, BSZ, 256, UU, 1, 1);

        // logits = h1 @ W2^T + b2  (64 x 5000, K=256)
        gemm(h1_bf, 256, 0, W2_bf, 256, 0, b2,
             logits, VOC, 0, nullptr, 0, 0, BSZ, VOC, 256, 0, 1);

        k_softmax<<<BSZ, 256, 0, stream>>>(logits, out_probs, t);
    }
}
